// VisionMinLSTM_523986010159
// MI455X (gfx1250) — matrix operand order repaired and checked by reading, not compiled, not run on hardware
//
#include <hip/hip_runtime.h>
#include <hip/hip_bf16.h>

typedef __attribute__((ext_vector_type(16))) __bf16 v16bf;
typedef __attribute__((ext_vector_type(8)))  __bf16 v8bf;
typedef __attribute__((ext_vector_type(8)))  float  v8f;

#define SEQ     196
#define NBATCH  128
#define DM      192
#define ROWS    (NBATCH*SEQ)      // 25088
#define NDEPTH  12
#define NCLS    1000
#define NCLSP   1024              // padded head rows
#define KPATCH  768
#define KHEAD   384
#define NCAT    576
#define IMG_N   ((size_t)NBATCH*3*224*224)   // 19,267,584

// WMMA GEMM tiling
#define BM 128
#define BN 64
#define BK 32
#define LPAD 40                   // LDS row stride in bf16 (padded, 80B = 5x16B)

// ---- workspace layout (bytes) ----
#define XS_OFF    ((size_t)0)
#define XS_BYTES  ((size_t)ROWS*DM*4)
#define LIN_OFF   (XS_OFF + XS_BYTES)
#define LIN_BYTES ((size_t)ROWS*NCAT*4)
#define LOGF_OFF  (LIN_OFF + LIN_BYTES)
#define G_BYTES   ((size_t)ROWS*DM*4)
#define V_OFF     (LOGF_OFF + G_BYTES)
#define XSBF_OFF  (V_OFF + G_BYTES)
#define PWB_OFF   (XSBF_OFF + (size_t)ROWS*DM*2)
#define WCAT_OFF  (PWB_OFF + (size_t)DM*KPATCH*2)
#define HWB_OFF   (WCAT_OFF + (size_t)NDEPTH*NCAT*DM*2)
// aliases (time-disjoint users of the big LIN region)
#define XBF_OFF   LIN_OFF         // bf16 input image, pre-layer phase only
#define HSEQ_OFF  LIN_OFF         // hseq, after gate has consumed lin
#define POOL_OFF  LOGF_OFF        // bf16 pooled vector, after last layer

// ---------------- async global->LDS (guarded; falls back to sync copy) ----
#if defined(__has_builtin)
#if __has_builtin(__builtin_amdgcn_global_load_async_to_lds_b128) && \
    __has_builtin(__builtin_amdgcn_s_wait_asynccnt)
#define HAS_ASYNC_LDS 1
#endif
#endif
#ifndef HAS_ASYNC_LDS
#define HAS_ASYNC_LDS 0
#endif

typedef __attribute__((__vector_size__(16))) int i32x4;

static __device__ __forceinline__ void cp16(__bf16* dst, const __bf16* src) {
#if HAS_ASYNC_LDS
  __builtin_amdgcn_global_load_async_to_lds_b128(
      (__attribute__((address_space(1))) i32x4*)(i32x4*)(void*)const_cast<__bf16*>(src),
      (__attribute__((address_space(3))) i32x4*)(i32x4*)(void*)dst, 0, 0);
#else
  *(uint4*)dst = *(const uint4*)src;
#endif
}

#if HAS_ASYNC_LDS
#define WAIT_ASYNC(n) __builtin_amdgcn_s_wait_asynccnt(n)
#else
#define WAIT_ASYNC(n) do {} while (0)
#endif

// ---------------- math helpers ----------------
static __device__ __forceinline__ float sp(float x) {           // softplus, stable
  return fmaxf(x, 0.f) + log1pf(expf(-fabsf(x)));
}
static __device__ __forceinline__ float log_g(float x) {
  return (x >= 0.f) ? logf(x + 0.5f) : -sp(-x);
}

// ---------------- WMMA fragment loads (wave32, per ISA 16-bit layouts) -----
// A 16x32 (MxK): lane m=l&15, hi=l>>4; elem e: K = e + 8*hi + (e>=8 ? 8 : 0)
static __device__ __forceinline__ v16bf frag_a(const __bf16* row, int hi) {
  union { v16bf v; v8bf h[2]; } u;
  u.h[0] = *(const v8bf*)(row + 8 * hi);
  u.h[1] = *(const v8bf*)(row + 16 + 8 * hi);
  return u.v;
}
// B 32x16 (KxN): lane n=l&15, hi=l>>4; elem e: K = e + 8*hi + (e>=8 ? 8 : 0)
static __device__ __forceinline__ v16bf frag_b(const __bf16* row, int hi) {
  union { v16bf v; v8bf h[2]; } u;
  u.h[0] = *(const v8bf*)(row + 8 * hi);
  u.h[1] = *(const v8bf*)(row + 16 + 8 * hi);
  return u.v;
}

static __device__ __forceinline__ void mma_step(
    const __bf16 (&As)[BM][LPAD], const __bf16 (&Bs)[BN][LPAD],
    v8f (&acc)[2][2], int wm, int wn, int lane) {
  const int lm = lane & 15, hi = lane >> 4;
  v16bf a0 = frag_a(&As[wm * 32 + lm][0], hi);
  v16bf a1 = frag_a(&As[wm * 32 + 16 + lm][0], hi);
  v16bf b0 = frag_b(&Bs[wn * 32 + lm][0], hi);
  v16bf b1 = frag_b(&Bs[wn * 32 + 16 + lm][0], hi);
  acc[0][0] = __builtin_amdgcn_wmma_f32_16x16x32_bf16(false, a0, false, b0, (short)0, acc[0][0], false, false);
  acc[0][1] = __builtin_amdgcn_wmma_f32_16x16x32_bf16(false, a0, false, b1, (short)0, acc[0][1], false, false);
  acc[1][0] = __builtin_amdgcn_wmma_f32_16x16x32_bf16(false, a1, false, b0, (short)0, acc[1][0], false, false);
  acc[1][1] = __builtin_amdgcn_wmma_f32_16x16x32_bf16(false, a1, false, b1, (short)0, acc[1][1], false, false);
}

// Double-buffered K pipeline: async-prefetch tile ks+1 while WMMAing tile ks.
// fa(ks, i): global src for A chunk i (i=0 -> row t>>2, i=1 -> row t>>2+64)
// fb(ks):    global src for B chunk
template <class FA, class FB>
static __device__ __forceinline__ void run_pipeline(
    int ksteps, FA fa, FB fb,
    __bf16 (&As)[2][BM][LPAD], __bf16 (&Bs)[2][BN][LPAD],
    v8f (&acc)[2][2], int t) {
  const int lane = t & 31, wave = t >> 5;
  const int wm = wave & 3, wn = wave >> 2;
  const int a_row = t >> 2, a_col = (t & 3) * 8;  // 128x32 tile = 512 16B chunks
  const int b_row = t >> 2, b_col = (t & 3) * 8;  // 64x32 tile = 256 chunks

  auto stage = [&](int ks, int buf) {
    cp16(&As[buf][a_row][a_col],      fa(ks, 0));
    cp16(&As[buf][a_row + 64][a_col], fa(ks, 1));
    cp16(&Bs[buf][b_row][b_col],      fb(ks));
  };

  stage(0, 0);
  for (int ks = 0; ks < ksteps; ++ks) {
    const int cur = ks & 1;
    if (ks + 1 < ksteps) {
      stage(ks + 1, cur ^ 1);
      WAIT_ASYNC(3);              // 3 newest in flight; tile-ks loads done
    } else {
      WAIT_ASYNC(0);
    }
    __syncthreads();              // everyone's tile-ks data visible in LDS
    mma_step(As[cur], Bs[cur], acc, wm, wn, lane);
    __syncthreads();              // buf consumed before it is re-staged
  }
}

// ---------------- prep kernels ----------------
__global__ void cvt_bf16_kernel(const float* __restrict__ s, __bf16* __restrict__ d, size_t n) {
  size_t i = (size_t)blockIdx.x * 256 + threadIdx.x;
  if (i < n) d[i] = (__bf16)s[i];
}

__global__ void zero_bf16_kernel(__bf16* __restrict__ d, int n) {
  int i = blockIdx.x * 256 + threadIdx.x;
  if (i < n) d[i] = (__bf16)0.f;
}

// pack wf/wi/wh into (NDEPTH, 576, 192) bf16
__global__ void wcat_kernel(const float* __restrict__ wf, const float* __restrict__ wi,
                            const float* __restrict__ wh, __bf16* __restrict__ wcat, int total) {
  int i = blockIdx.x * 256 + threadIdx.x;
  if (i >= total) return;
  int d = i % DM;
  int r = (i / DM) % NCAT;
  int l = i / (DM * NCAT);
  const float* src = wf; int rr = r;
  if (r >= 384)      { src = wh; rr = r - 384; }
  else if (r >= 192) { src = wi; rr = r - 192; }
  wcat[i] = (__bf16)src[((size_t)l * DM + rr) * DM + d];
}

// ---------------- patch-embed GEMM (im2col A from bf16 image) -------------
__global__ __launch_bounds__(256)
void patch_gemm_kernel(const __bf16* __restrict__ xbf, const __bf16* __restrict__ pwb,
                       const float* __restrict__ pb, const float* __restrict__ pos,
                       float* __restrict__ xs, __bf16* __restrict__ xsbf) {
  __shared__ __align__(16) __bf16 As[2][BM][LPAD];
  __shared__ __align__(16) __bf16 Bs[2][BN][LPAD];
  const int t = threadIdx.x;
  const int m0 = blockIdx.x * BM;
  const int n0 = blockIdx.y * BN;       // < 192
  v8f acc[2][2] = {};

  // hoist im2col row bases out of the K loop (no divisions inside)
  const int a_row = t >> 2, a_col = (t & 3) * 8;
  auto mkb = [&](int m) -> const __bf16* {
    int bb = m / SEQ, s = m - bb * SEQ;
    int ph = s / 14, pw_ = s - ph * 14;
    return xbf + ((size_t)bb * 3 * 224 + ph * 16) * 224 + pw_ * 16;
  };
  const __bf16* ab[2] = { mkb(m0 + a_row), mkb(m0 + a_row + 64) };
  const __bf16* bB = pwb + (size_t)(n0 + (t >> 2)) * KPATCH + ((t & 3) * 8);

  run_pipeline(KPATCH / BK,
    [&](int ks, int i) -> const __bf16* {
      int kk = ks * BK + a_col;
      int c = kk >> 8, rem = kk & 255, yy = rem >> 4, xx = rem & 15;
      return ab[i] + (size_t)c * (224 * 224) + yy * 224 + xx;
    },
    [&](int ks) -> const __bf16* { return bB + ks * BK; },
    As, Bs, acc, t);

  const int lane = t & 31, wave = t >> 5;
  const int wm = wave & 3, wn = wave >> 2;
  const int lm = lane & 15, hi = lane >> 4;
#pragma unroll
  for (int tn = 0; tn < 2; ++tn) {
    int n = n0 + wn * 32 + tn * 16 + lm;
    float bias = pb[n];
#pragma unroll
    for (int tm = 0; tm < 2; ++tm) {
      int mb = m0 + wm * 32 + tm * 16 + 8 * hi;
#pragma unroll
      for (int r = 0; r < 8; ++r) {
        int m = mb + r;
        int s = m % SEQ;
        float v = acc[tm][tn][r] + bias + pos[(size_t)s * DM + n];
        xs[(size_t)m * DM + n] = v;
        xsbf[(size_t)m * DM + n] = (__bf16)v;
      }
    }
  }
}

// ---------------- per-layer fused f/i/h GEMM ----------------
__global__ __launch_bounds__(256)
void lin_gemm_kernel(const __bf16* __restrict__ xsbf, const __bf16* __restrict__ wcat,
                     const float* __restrict__ bfv, const float* __restrict__ biv,
                     const float* __restrict__ bhv, float* __restrict__ lin) {
  __shared__ __align__(16) __bf16 As[2][BM][LPAD];
  __shared__ __align__(16) __bf16 Bs[2][BN][LPAD];
  const int t = threadIdx.x;
  const int m0 = blockIdx.x * BM;
  const int n0 = blockIdx.y * BN;       // < 576
  v8f acc[2][2] = {};

  const __bf16* aB0 = xsbf + (size_t)(m0 + (t >> 2)) * DM + ((t & 3) * 8);
  const __bf16* aB1 = aB0 + (size_t)64 * DM;
  const __bf16* bB  = wcat + (size_t)(n0 + (t >> 2)) * DM + ((t & 3) * 8);

  run_pipeline(DM / BK,
    [&](int ks, int i) -> const __bf16* { return (i ? aB1 : aB0) + ks * BK; },
    [&](int ks) -> const __bf16* { return bB + ks * BK; },
    As, Bs, acc, t);

  const int lane = t & 31, wave = t >> 5;
  const int wm = wave & 3, wn = wave >> 2;
  const int lm = lane & 15, hi = lane >> 4;
#pragma unroll
  for (int tn = 0; tn < 2; ++tn) {
    int n = n0 + wn * 32 + tn * 16 + lm;
    float bias = (n < 192) ? bfv[n] : (n < 384) ? biv[n - 192] : bhv[n - 384];
#pragma unroll
    for (int tm = 0; tm < 2; ++tm) {
      int mb = m0 + wm * 32 + tm * 16 + 8 * hi;
#pragma unroll
      for (int r = 0; r < 8; ++r)
        lin[(size_t)(mb + r) * NCAT + n] = acc[tm][tn][r] + bias;
    }
  }
}

// ---------------- gate: log_f and v from f/i/h linears ----------------
__global__ void gate_kernel(const float* __restrict__ lin, float* __restrict__ lf,
                            float* __restrict__ vout) {
  int i = blockIdx.x * 256 + threadIdx.x;
  if (i >= ROWS * DM) return;
  int row = i / DM, d = i - row * DM;
  const float* L = lin + (size_t)row * NCAT;
  float f = L[d], g = L[192 + d], h = L[384 + d];
  float diff = sp(-f) - sp(-g);
  float log_f = -sp(diff);
  float log_i = -sp(-diff);
  lf[i] = log_f;
  vout[i] = expf(log_i + log_g(h));
}

// ---------------- sequential minLSTM scan over S ----------------
__global__ void scan_kernel(const float* __restrict__ lf, const float* __restrict__ v,
                            float* __restrict__ hseq) {
  int i = blockIdx.x * 256 + threadIdx.x;     // 0 .. NBATCH*DM-1
  int b = i / DM, d = i - b * DM;
  float h = 0.f;
  size_t base = (size_t)b * SEQ * DM + d;
  for (int s = 0; s < SEQ; ++s) {
    size_t idx = base + (size_t)s * DM;
    h = expf(log_g(h) - lf[idx]) + v[idx];
    hseq[idx] = h;
  }
}

// ---------------- residual + LayerNorm (one wave32 per row) ----------------
__global__ __launch_bounds__(256)
void ln_kernel(const float* __restrict__ hseq, float* __restrict__ xs,
               __bf16* __restrict__ xsbf,
               const float* __restrict__ w, const float* __restrict__ b) {
  const int wave = threadIdx.x >> 5, lane = threadIdx.x & 31;
  const int row = blockIdx.x * 8 + wave;
  const float* hp = hseq + (size_t)row * DM;
  float* xp = xs + (size_t)row * DM;
  __bf16* xbp = xsbf + (size_t)row * DM;
  float vals[6];
  float sum = 0.f;
#pragma unroll
  for (int j = 0; j < 6; ++j) {
    int d = lane + j * 32;
    float t = hp[d] + xp[d];
    vals[j] = t; sum += t;
  }
#pragma unroll
  for (int m = 16; m >= 1; m >>= 1) sum += __shfl_xor(sum, m, 32);
  float mu = sum * (1.f / 192.f);
  float var = 0.f;
#pragma unroll
  for (int j = 0; j < 6; ++j) { float c = vals[j] - mu; var += c * c; }
#pragma unroll
  for (int m = 16; m >= 1; m >>= 1) var += __shfl_xor(var, m, 32);
  float rs = rsqrtf(var * (1.f / 192.f) + 1e-5f);
#pragma unroll
  for (int j = 0; j < 6; ++j) {
    int d = lane + j * 32;
    float y = (vals[j] - mu) * rs * w[d] + b[d];
    xp[d] = y;
    xbp[d] = (__bf16)y;
  }
}

// ---------------- pooling (bf16) + head GEMM ----------------
__global__ void pool_kernel(const float* __restrict__ xs, __bf16* __restrict__ pooled) {
  int i = blockIdx.x * 256 + threadIdx.x;
  if (i >= NBATCH * 2 * DM) return;
  int bb = i / (2 * DM), j = i - bb * 2 * DM;
  float v = (j < DM) ? xs[(size_t)bb * SEQ * DM + j]
                     : xs[((size_t)bb * SEQ + SEQ - 1) * DM + (j - DM)];
  pooled[i] = (__bf16)v;
}

__global__ __launch_bounds__(256)
void head_gemm_kernel(const __bf16* __restrict__ pooled, const __bf16* __restrict__ hwb,
                      const float* __restrict__ hb, float* __restrict__ out) {
  __shared__ __align__(16) __bf16 As[2][BM][LPAD];
  __shared__ __align__(16) __bf16 Bs[2][BN][LPAD];
  const int t = threadIdx.x;
  const int n0 = blockIdx.y * BN;       // hwb padded to 1024 rows, zero-filled
  v8f acc[2][2] = {};

  const __bf16* aB0 = pooled + (size_t)(t >> 2) * KHEAD + ((t & 3) * 8);
  const __bf16* aB1 = aB0 + (size_t)64 * KHEAD;
  const __bf16* bB  = hwb + (size_t)(n0 + (t >> 2)) * KHEAD + ((t & 3) * 8);

  run_pipeline(KHEAD / BK,
    [&](int ks, int i) -> const __bf16* { return (i ? aB1 : aB0) + ks * BK; },
    [&](int ks) -> const __bf16* { return bB + ks * BK; },
    As, Bs, acc, t);

  const int lane = t & 31, wave = t >> 5;
  const int wm = wave & 3, wn = wave >> 2;
  const int lm = lane & 15, hi = lane >> 4;
#pragma unroll
  for (int tn = 0; tn < 2; ++tn) {
    int n = n0 + wn * 32 + tn * 16 + lm;
    if (n >= NCLS) continue;
    float bias = hb[n];
#pragma unroll
    for (int tm = 0; tm < 2; ++tm) {
      int mb = wm * 32 + tm * 16 + 8 * hi;
#pragma unroll
      for (int r = 0; r < 8; ++r)
        out[(size_t)(mb + r) * NCLS + n] = acc[tm][tn][r] + bias;
    }
  }
}

// ---------------- host orchestration ----------------
extern "C" void kernel_launch(void* const* d_in, const int* in_sizes, int n_in,
                              void* d_out, int out_size, void* d_ws, size_t ws_size,
                              hipStream_t stream) {
  (void)in_sizes; (void)n_in; (void)out_size; (void)ws_size;
  const float* x       = (const float*)d_in[0];
  const float* patch_w = (const float*)d_in[1];
  const float* patch_b = (const float*)d_in[2];
  const float* pos     = (const float*)d_in[3];
  const float* wf      = (const float*)d_in[4];
  const float* bfv     = (const float*)d_in[5];
  const float* wi      = (const float*)d_in[6];
  const float* biv     = (const float*)d_in[7];
  const float* wh      = (const float*)d_in[8];
  const float* bhv     = (const float*)d_in[9];
  const float* nw      = (const float*)d_in[10];
  const float* nb      = (const float*)d_in[11];
  const float* head_w  = (const float*)d_in[12];
  const float* head_b  = (const float*)d_in[13];
  float* out = (float*)d_out;

  char* ws = (char*)d_ws;
  float*  xs     = (float*)(ws + XS_OFF);
  float*  lin    = (float*)(ws + LIN_OFF);
  float*  lf     = (float*)(ws + LOGF_OFF);
  float*  vbuf   = (float*)(ws + V_OFF);
  float*  hseq   = (float*)(ws + HSEQ_OFF);     // aliases lin (dead after gate)
  __bf16* xbf    = (__bf16*)(ws + XBF_OFF);     // aliases lin (pre-layer phase)
  __bf16* pooled = (__bf16*)(ws + POOL_OFF);    // aliases lf (post layers)
  __bf16* xsbf   = (__bf16*)(ws + XSBF_OFF);
  __bf16* pwb    = (__bf16*)(ws + PWB_OFF);
  __bf16* wcat   = (__bf16*)(ws + WCAT_OFF);
  __bf16* hwb    = (__bf16*)(ws + HWB_OFF);

  // prep: bf16 image + bf16 weights (padded head)
  cvt_bf16_kernel<<<(unsigned)((IMG_N + 255) / 256), 256, 0, stream>>>(x, xbf, IMG_N);
  cvt_bf16_kernel<<<(DM * KPATCH + 255) / 256, 256, 0, stream>>>(patch_w, pwb, (size_t)DM * KPATCH);
  wcat_kernel<<<(NDEPTH * NCAT * DM + 255) / 256, 256, 0, stream>>>(wf, wi, wh, wcat, NDEPTH * NCAT * DM);
  cvt_bf16_kernel<<<(NCLS * KHEAD + 255) / 256, 256, 0, stream>>>(head_w, hwb, (size_t)NCLS * KHEAD);
  zero_bf16_kernel<<<((NCLSP - NCLS) * KHEAD + 255) / 256, 256, 0, stream>>>(
      hwb + (size_t)NCLS * KHEAD, (NCLSP - NCLS) * KHEAD);

  // patch embed + pos embed (writes xs fp32 + xs_bf mirror)
  patch_gemm_kernel<<<dim3(ROWS / BM, DM / BN), 256, 0, stream>>>(xbf, pwb, patch_b, pos, xs, xsbf);

  for (int l = 0; l < NDEPTH; ++l) {
    lin_gemm_kernel<<<dim3(ROWS / BM, NCAT / BN), 256, 0, stream>>>(
        xsbf, wcat + (size_t)l * NCAT * DM, bfv + l * DM, biv + l * DM, bhv + l * DM, lin);
    gate_kernel<<<(ROWS * DM + 255) / 256, 256, 0, stream>>>(lin, lf, vbuf);
    scan_kernel<<<(NBATCH * DM) / 256, 256, 0, stream>>>(lf, vbuf, hseq);
    ln_kernel<<<ROWS / 8, 256, 0, stream>>>(hseq, xs, xsbf, nw, nb);
  }

  pool_kernel<<<(NBATCH * 2 * DM + 255) / 256, 256, 0, stream>>>(xs, pooled);
  head_gemm_kernel<<<dim3(1, NCLSP / BN), 256, 0, stream>>>(pooled, hwb, head_b, out);
}
